// CSNA_4337916969343
// MI455X (gfx1250) — matrix operand order repaired and checked by reading, not compiled, not run on hardware
//
#include <hip/hip_runtime.h>
#include <hip/hip_bf16.h>

// ---------------------------------------------------------------------------
// CSNA GNN inference for gfx1250 (MI455X). GEMMs via v_wmma_f32_16x16x32_bf16,
// segment-softmax message passing via hardware f32 global atomics.
// ---------------------------------------------------------------------------

typedef __attribute__((ext_vector_type(16))) __bf16 v16bf;
typedef __attribute__((ext_vector_type(8)))  float  v8f;

#define N_NODES 50000
#define N_EDGES 400000
#define ETOT    450000   // edges + self loops
#define F_INDIM 500
#define KPAD    512      // F_IN padded to multiple of 32
#define HDIM    128
#define NCLS    40
#define CPAD    48       // classes padded to multiple of 16
#define BN_EPS  1e-5f

// ---- helpers ---------------------------------------------------------------

__device__ __forceinline__ unsigned short f2bf(float f) {
  unsigned int u = __float_as_uint(f);
  u += 0x7fffu + ((u >> 16) & 1u);          // round-to-nearest-even
  return (unsigned short)(u >> 16);
}

// A-matrix (16x32 bf16) per-lane fragment: lane m=lane&15, half=lane>>4.
// ISA layout: elements 0..7  = A[m, kb + 8*half .. +7]
//             elements 8..15 = A[m, kb + 16 + 8*half .. +7]
__device__ __forceinline__ v16bf load_a16(const unsigned short* row, int kb, int half) {
  union { uint4 u[2]; v16bf v; } t;
  t.u[0] = *(const uint4*)(row + kb + 8 * half);
  t.u[1] = *(const uint4*)(row + kb + 16 + 8 * half);
  return t.v;
}

// B-matrix (32x16 bf16) per-lane fragment: lane n=lane&15, half=lane>>4.
// ISA layout: elements 0..7 = W[j_n, kb + 8*half .. +7], elements 8..15 = W[j_n, kb + 16 + 8*half .. +7]
__device__ __forceinline__ v16bf load_b16(const unsigned short* wrow, int kb, int half) {
  union { uint4 u[2]; v16bf v; } t;
  const unsigned short* p = wrow + kb + 8 * half;
  t.u[0] = *(const uint4*)(p);
  t.u[1] = *(const uint4*)(p + 16);
  return t.v;
}

__device__ __forceinline__ float waveReduceSum(float v) {
  #pragma unroll
  for (int off = 16; off > 0; off >>= 1) v += __shfl_xor(v, off, 32);
  return v;
}

// ---- weight packing --------------------------------------------------------

__global__ void kPackMlpW(const float* __restrict__ W, unsigned short* __restrict__ out) {
  int i = blockIdx.x * blockDim.x + threadIdx.x;      // HDIM*KPAD
  if (i >= HDIM * KPAD) return;
  int j = i / KPAD, k = i % KPAD;
  out[i] = f2bf(k < F_INDIM ? W[j * F_INDIM + k] : 0.f);
}

// Wall[l][512][128] = rows [Wg | Wcon | Wdis | Wself] of layer l
__global__ void kPackWall(const float* __restrict__ Wg, const float* __restrict__ Wcon,
                          const float* __restrict__ Wdis, const float* __restrict__ Wself,
                          unsigned short* __restrict__ out) {
  int i = blockIdx.x * blockDim.x + threadIdx.x;      // 2*512*128
  if (i >= 2 * 4 * HDIM * HDIM) return;
  int l   = i / (4 * HDIM * HDIM);
  int r   = (i / HDIM) % (4 * HDIM);
  int k   = i % HDIM;
  int blk = r / HDIM, rr = r % HDIM;
  const float* W = (blk == 0) ? Wg : (blk == 1) ? Wcon : (blk == 2) ? Wdis : Wself;
  out[i] = f2bf(W[(l * HDIM + rr) * HDIM + k]);
}

__global__ void kPackClfW(const float* __restrict__ W, unsigned short* __restrict__ out) {
  int i = blockIdx.x * blockDim.x + threadIdx.x;      // CPAD*HDIM
  if (i >= CPAD * HDIM) return;
  int j = i / HDIM, k = i % HDIM;
  out[i] = f2bf(j < NCLS ? W[j * HDIM + k] : 0.f);
}

__global__ void kConvX(const float* __restrict__ x, unsigned short* __restrict__ xbf) {
  int i = blockIdx.x * blockDim.x + threadIdx.x;      // N*KPAD
  if (i >= N_NODES * KPAD) return;
  int n = i / KPAD, k = i % KPAD;
  xbf[i] = f2bf(k < F_INDIM ? x[(long)n * F_INDIM + k] : 0.f);
}

__global__ void kZero(float* __restrict__ p, int n) {
  int i = blockIdx.x * blockDim.x + threadIdx.x;
  if (i < n) p[i] = 0.f;
}

// ---- input MLP: h = relu(BN(xbf @ mlpW^T + b)) -----------------------------
// block = 128 threads (4 waves); block handles 16 rows, wave w -> cols [32w,32w+32)
__global__ void kGemmMlp(const unsigned short* __restrict__ xbf,
                         const unsigned short* __restrict__ wbf,
                         const float* __restrict__ bias,
                         const float* __restrict__ gamma, const float* __restrict__ beta,
                         const float* __restrict__ rm, const float* __restrict__ rv,
                         float* __restrict__ h, unsigned short* __restrict__ hbf) {
  int wave = threadIdx.x >> 5;
  int lane = threadIdx.x & 31;
  int half = lane >> 4, ln = lane & 15;
  int mbase = blockIdx.x * 16;
  int m = mbase + ln; if (m >= N_NODES) m = N_NODES - 1;
  const unsigned short* arow = xbf + (long)m * KPAD;
  int jb = wave * 32;
  v8f c0 = {}, c1 = {};
  for (int kb = 0; kb < KPAD; kb += 32) {
    v16bf a  = load_a16(arow, kb, half);
    v16bf b0 = load_b16(wbf + (long)(jb + ln) * KPAD, kb, half);
    v16bf b1 = load_b16(wbf + (long)(jb + 16 + ln) * KPAD, kb, half);
    c0 = __builtin_amdgcn_wmma_f32_16x16x32_bf16(false, a, false, b0, (short)0, c0, false, false);
    c1 = __builtin_amdgcn_wmma_f32_16x16x32_bf16(false, a, false, b1, (short)0, c1, false, false);
  }
  #pragma unroll
  for (int r = 0; r < 8; ++r) {
    int row = mbase + r + 8 * half;
    if (row >= N_NODES) continue;
    #pragma unroll
    for (int t = 0; t < 2; ++t) {
      int col = jb + t * 16 + ln;
      float y = (t == 0) ? c0[r] : c1[r];
      y += bias[col];
      y = (y - rm[col]) * rsqrtf(rv[col] + BN_EPS) * gamma[col] + beta[col];
      y = fmaxf(y, 0.f);
      h  [(long)row * HDIM + col] = y;
      hbf[(long)row * HDIM + col] = f2bf(y);
    }
  }
}

// ---- fused per-layer GEMM: z[N,512] = hbf @ [Wg|Wcon|Wdis|Wself]^T ---------
// block = 256 threads (8 waves); block handles 16 rows, wave w -> cols [64w,64w+64)
__global__ void kGemmAll(const unsigned short* __restrict__ hbf,
                         const unsigned short* __restrict__ wall,
                         const float* __restrict__ bself,
                         float* __restrict__ z) {
  int wave = threadIdx.x >> 5;
  int lane = threadIdx.x & 31;
  int half = lane >> 4, ln = lane & 15;
  int mbase = blockIdx.x * 16;
  int m = mbase + ln; if (m >= N_NODES) m = N_NODES - 1;
  const unsigned short* arow = hbf + (long)m * HDIM;
  int jb = wave * 64;
  v8f c[4] = {};
  for (int kb = 0; kb < HDIM; kb += 32) {
    v16bf a = load_a16(arow, kb, half);
    #pragma unroll
    for (int t = 0; t < 4; ++t) {
      v16bf b = load_b16(wall + (long)(jb + t * 16 + ln) * HDIM, kb, half);
      c[t] = __builtin_amdgcn_wmma_f32_16x16x32_bf16(false, a, false, b, (short)0, c[t], false, false);
    }
  }
  #pragma unroll
  for (int r = 0; r < 8; ++r) {
    int row = mbase + r + 8 * half;
    if (row >= N_NODES) continue;
    #pragma unroll
    for (int t = 0; t < 4; ++t) {
      int col = jb + t * 16 + ln;
      float y = c[t][r];
      if (col >= 384) y += bself[col - 384];      // self branch bias
      z[(long)row * KPAD + col] = y;
    }
  }
}

// ---- edge pass 1: scores + softmax denominators ----------------------------
// one wave per edge; xg = z[:,0:128]
__global__ void kEdge1(const int* __restrict__ ei, const float* __restrict__ z,
                       float* __restrict__ econ, float* __restrict__ edis,
                       float* __restrict__ sum_con, float* __restrict__ sum_dis) {
  int gw   = (blockIdx.x * blockDim.x + threadIdx.x) >> 5;
  int lane = threadIdx.x & 31;
  if (gw >= ETOT) return;
  int row, col;
  if (gw < N_EDGES) { row = ei[gw]; col = ei[N_EDGES + gw]; }
  else              { row = gw - N_EDGES; col = row; }
  float4 a = *(const float4*)(z + (long)row * KPAD + lane * 4);
  float4 b = *(const float4*)(z + (long)col * KPAD + lane * 4);
  float dx = a.x - b.x, dy = a.y - b.y, dz = a.z - b.z, dw = a.w - b.w;
  float d = waveReduceSum(dx * dx + dy * dy + dz * dz + dw * dw);
  if (lane == 0) {
    float g = sqrtf(fmaxf(d, 1e-24f));
    float s = 1.f / (1.f + expf(g));              // sigmoid(-g/TAU), TAU=1
    float ec = expf(s);
    float ed = expf(1.f - s);
    econ[gw] = ec; edis[gw] = ed;
    unsafeAtomicAdd(sum_con + row, ec);
    unsafeAtomicAdd(sum_dis + row, ed);
  }
}

// ---- edge pass 2: weighted scatter of neighbor features --------------------
// hcon = z[:,128:256], hdis = z[:,256:384]
__global__ void kEdge2(const int* __restrict__ ei, const float* __restrict__ z,
                       const float* __restrict__ econ, const float* __restrict__ edis,
                       const float* __restrict__ sum_con, const float* __restrict__ sum_dis,
                       float* __restrict__ out_con, float* __restrict__ out_dis) {
  int gw   = (blockIdx.x * blockDim.x + threadIdx.x) >> 5;
  int lane = threadIdx.x & 31;
  if (gw >= ETOT) return;
  int row, col;
  if (gw < N_EDGES) { row = ei[gw]; col = ei[N_EDGES + gw]; }
  else              { row = gw - N_EDGES; col = row; }
  float wc = econ[gw] / (sum_con[row] + 1e-16f);
  float wd = edis[gw] / (sum_dis[row] + 1e-16f);
  float4 c4 = *(const float4*)(z + (long)col * KPAD + 128 + lane * 4);
  float4 d4 = *(const float4*)(z + (long)col * KPAD + 256 + lane * 4);
  float* oc = out_con + (long)row * HDIM + lane * 4;
  float* od = out_dis + (long)row * HDIM + lane * 4;
  unsafeAtomicAdd(oc + 0, wc * c4.x);
  unsafeAtomicAdd(oc + 1, wc * c4.y);
  unsafeAtomicAdd(oc + 2, wc * c4.z);
  unsafeAtomicAdd(oc + 3, wc * c4.w);
  unsafeAtomicAdd(od + 0, wd * d4.x);
  unsafeAtomicAdd(od + 1, wd * d4.y);
  unsafeAtomicAdd(od + 2, wd * d4.z);
  unsafeAtomicAdd(od + 3, wd * d4.w);
}

// ---- gate + combine + (BN/ReLU) + residual ---------------------------------
// one wave per node, each lane owns 4 features
__global__ void kCombine(const float* __restrict__ out_con, const float* __restrict__ out_dis,
                         const float* __restrict__ z,
                         const float* __restrict__ Wgate, const float* __restrict__ bgate,
                         const float* __restrict__ gamma, const float* __restrict__ beta,
                         const float* __restrict__ rm, const float* __restrict__ rv,
                         int use_bn,
                         float* __restrict__ h, unsigned short* __restrict__ hbf) {
  int node = (blockIdx.x * blockDim.x + threadIdx.x) >> 5;
  int lane = threadIdx.x & 31;
  if (node >= N_NODES) return;
  int f = lane * 4;
  float4 con = *(const float4*)(out_con + (long)node * HDIM + f);
  float4 dis = *(const float4*)(out_dis + (long)node * HDIM + f);
  float4 slf = *(const float4*)(z + (long)node * KPAD + 384 + f);
  float logit[3];
  #pragma unroll
  for (int r = 0; r < 3; ++r) {
    const float* wr = Wgate + r * 384;
    float p = wr[f] * con.x + wr[f + 1] * con.y + wr[f + 2] * con.z + wr[f + 3] * con.w;
    const float* wrd = wr + 128;
    p += wrd[f] * dis.x + wrd[f + 1] * dis.y + wrd[f + 2] * dis.z + wrd[f + 3] * dis.w;
    const float* wrs = wr + 256;
    p += wrs[f] * slf.x + wrs[f + 1] * slf.y + wrs[f + 2] * slf.z + wrs[f + 3] * slf.w;
    logit[r] = waveReduceSum(p) + bgate[r];
  }
  float mx = fmaxf(logit[0], fmaxf(logit[1], logit[2]));
  float e0 = expf(logit[0] - mx), e1 = expf(logit[1] - mx), e2 = expf(logit[2] - mx);
  float inv = 1.f / (e0 + e1 + e2);
  float g0 = e0 * inv, g1 = e1 * inv, g2 = e2 * inv;
  float o[4];
  o[0] = g0 * con.x + g1 * dis.x + g2 * slf.x;
  o[1] = g0 * con.y + g1 * dis.y + g2 * slf.y;
  o[2] = g0 * con.z + g1 * dis.z + g2 * slf.z;
  o[3] = g0 * con.w + g1 * dis.w + g2 * slf.w;
  long base = (long)node * HDIM + f;
  #pragma unroll
  for (int i = 0; i < 4; ++i) {
    float y = o[i];
    int col = f + i;
    if (use_bn) {
      y = (y - rm[col]) * rsqrtf(rv[col] + BN_EPS) * gamma[col] + beta[col];
      y = fmaxf(y, 0.f);
    }
    y += h[base + i];                 // residual with layer input
    h[base + i] = y;
    hbf[base + i] = f2bf(y);
  }
}

// ---- classifier: out[N,40] = hbf @ clfW^T + b ------------------------------
// block = 128 threads (4 waves); wave w handles rows [64*bid + 16w, +16), 48 cols
__global__ void kGemmClf(const unsigned short* __restrict__ hbf,
                         const unsigned short* __restrict__ wbf,
                         const float* __restrict__ bias, float* __restrict__ out) {
  int wave = threadIdx.x >> 5;
  int lane = threadIdx.x & 31;
  int half = lane >> 4, ln = lane & 15;
  int mbase = blockIdx.x * 64 + wave * 16;
  if (mbase >= N_NODES) return;
  int m = mbase + ln; if (m >= N_NODES) m = N_NODES - 1;
  const unsigned short* arow = hbf + (long)m * HDIM;
  v8f c[3] = {};
  for (int kb = 0; kb < HDIM; kb += 32) {
    v16bf a = load_a16(arow, kb, half);
    #pragma unroll
    for (int t = 0; t < 3; ++t) {
      v16bf b = load_b16(wbf + (long)(t * 16 + ln) * HDIM, kb, half);
      c[t] = __builtin_amdgcn_wmma_f32_16x16x32_bf16(false, a, false, b, (short)0, c[t], false, false);
    }
  }
  #pragma unroll
  for (int r = 0; r < 8; ++r) {
    int row = mbase + r + 8 * half;
    if (row >= N_NODES) continue;
    #pragma unroll
    for (int t = 0; t < 3; ++t) {
      int col = t * 16 + ln;
      if (col < NCLS) out[(long)row * NCLS + col] = c[t][r] + bias[col];
    }
  }
}

// ---------------------------------------------------------------------------

extern "C" void kernel_launch(void* const* d_in, const int* in_sizes, int n_in,
                              void* d_out, int out_size, void* d_ws, size_t ws_size,
                              hipStream_t stream) {
  const float* x        = (const float*)d_in[0];
  const int*   ei       = (const int*)  d_in[1];
  const float* mlp_W    = (const float*)d_in[2];
  const float* mlp_b    = (const float*)d_in[3];
  const float* bn_in_g  = (const float*)d_in[4];
  const float* bn_in_b  = (const float*)d_in[5];
  const float* bn_in_rm = (const float*)d_in[6];
  const float* bn_in_rv = (const float*)d_in[7];
  const float* Wg       = (const float*)d_in[8];
  const float* Wcon     = (const float*)d_in[9];
  const float* Wdis     = (const float*)d_in[10];
  const float* Wself    = (const float*)d_in[11];
  const float* bself    = (const float*)d_in[12];
  const float* Wgate    = (const float*)d_in[13];
  const float* bgate    = (const float*)d_in[14];
  const float* bn_m_g   = (const float*)d_in[15];
  const float* bn_m_b   = (const float*)d_in[16];
  const float* bn_m_rm  = (const float*)d_in[17];
  const float* bn_m_rv  = (const float*)d_in[18];
  const float* clf_W    = (const float*)d_in[19];
  const float* clf_b    = (const float*)d_in[20];

  char*  ws  = (char*)d_ws;
  size_t off = 0;
  auto take = [&](size_t bytes) {
    char* p = ws + off;
    off = (off + bytes + 255) & ~(size_t)255;
    return p;
  };
  unsigned short* xbf     = (unsigned short*)take((size_t)N_NODES * KPAD * 2);
  unsigned short* mlpWbf  = (unsigned short*)take((size_t)HDIM * KPAD * 2);
  unsigned short* wallBf  = (unsigned short*)take((size_t)2 * 4 * HDIM * HDIM * 2);
  unsigned short* clfWbf  = (unsigned short*)take((size_t)CPAD * HDIM * 2);
  float*          hbuf    = (float*)         take((size_t)N_NODES * HDIM * 4);
  unsigned short* hbf     = (unsigned short*)take((size_t)N_NODES * HDIM * 2);
  float*          z       = (float*)         take((size_t)N_NODES * KPAD * 4);
  float*          econ    = (float*)         take((size_t)ETOT * 4);
  float*          edis    = (float*)         take((size_t)ETOT * 4);
  float*          out_con = (float*)         take((size_t)N_NODES * HDIM * 4);
  float*          out_dis = (float*)         take((size_t)N_NODES * HDIM * 4);
  float*          sum_con = (float*)         take((size_t)N_NODES * 4);
  float*          sum_dis = (float*)         take((size_t)N_NODES * 4);

  // --- pack weights / inputs to bf16 ---
  kPackMlpW<<<(HDIM * KPAD + 255) / 256, 256, 0, stream>>>(mlp_W, mlpWbf);
  kPackWall<<<(2 * 4 * HDIM * HDIM + 255) / 256, 256, 0, stream>>>(Wg, Wcon, Wdis, Wself, wallBf);
  kPackClfW<<<(CPAD * HDIM + 255) / 256, 256, 0, stream>>>(clf_W, clfWbf);
  kConvX<<<(N_NODES * KPAD + 255) / 256, 256, 0, stream>>>(x, xbf);

  // --- input MLP + BN + ReLU ---
  kGemmMlp<<<N_NODES / 16, 128, 0, stream>>>(xbf, mlpWbf, mlp_b, bn_in_g, bn_in_b,
                                             bn_in_rm, bn_in_rv, hbuf, hbf);

  // --- two CSNA conv layers ---
  for (int l = 0; l < 2; ++l) {
    kGemmAll<<<N_NODES / 16, 256, 0, stream>>>(hbf, wallBf + (size_t)l * 4 * HDIM * HDIM,
                                               bself + l * HDIM, z);
    kZero<<<(N_NODES * HDIM + 255) / 256, 256, 0, stream>>>(out_con, N_NODES * HDIM);
    kZero<<<(N_NODES * HDIM + 255) / 256, 256, 0, stream>>>(out_dis, N_NODES * HDIM);
    kZero<<<(N_NODES + 255) / 256, 256, 0, stream>>>(sum_con, N_NODES);
    kZero<<<(N_NODES + 255) / 256, 256, 0, stream>>>(sum_dis, N_NODES);
    kEdge1<<<(ETOT * 32 + 255) / 256, 256, 0, stream>>>(ei, z, econ, edis, sum_con, sum_dis);
    kEdge2<<<(ETOT * 32 + 255) / 256, 256, 0, stream>>>(ei, z, econ, edis, sum_con, sum_dis,
                                                        out_con, out_dis);
    kCombine<<<(N_NODES * 32 + 255) / 256, 256, 0, stream>>>(
        out_con, out_dis, z, Wgate + l * 3 * 384, bgate + l * 3,
        bn_m_g, bn_m_b, bn_m_rm, bn_m_rv, (l == 0) ? 1 : 0, hbuf, hbf);
  }

  // --- classifier ---
  kGemmClf<<<(N_NODES + 63) / 64, 128, 0, stream>>>(hbf, clfWbf, clf_b, (float*)d_out);
}
